// SS2D_55886114456130
// MI455X (gfx1250) — compile-verified, matrix operand order checked by reading, not run on hardware
//
#include <hip/hip_runtime.h>
#include <cstdint>
#include <cstddef>

// ---------------------------------------------------------------------------
// SS2D fused pipeline for gfx1250 (MI455X).
//   K0: convert + swizzle w_in / w_out to bf16 fragment-native layout
//   K1: x_proj GEMM (v_wmma_f32_16x16x32_bf16), TDM double-buffered B slabs,
//       LDS-transposed coalesced epilogue -> x_inner, silu(z)
//   K2: fused dw-conv3x3 + SSM + LayerNorm + out-GEMM (WMMA), TDM halo loads
// ---------------------------------------------------------------------------

typedef __bf16 bf16;
typedef __attribute__((ext_vector_type(16))) __bf16 v16bf;
typedef __attribute__((ext_vector_type(8)))  __bf16 v8bf;
typedef __attribute__((ext_vector_type(8)))  float  v8f;
typedef __attribute__((ext_vector_type(4)))  unsigned int v4u;
typedef __attribute__((ext_vector_type(4)))  int v4i;
typedef __attribute__((ext_vector_type(8)))  int v8i;

#define DEV __device__ __forceinline__

constexpr int BN   = 8;
constexpr int DM   = 192;
constexpr int DI   = 384;
constexpr int DS   = 16;
constexpr int HH   = 128;
constexpr int WWI  = 128;
constexpr int LL   = HH * WWI;     // 16384
constexpr int NROW = BN * LL;      // 131072
constexpr int N2   = 2 * DI;       // 768

DEV float siluf(float x)     { return x / (1.f + __expf(-x)); }
DEV float softplusf(float x) { return (x > 20.f) ? x : log1pf(__expf(x)); }

// Fragment layout (ISA 7.12.2, 16-bit A/B): element j of v16bf holds
// k = 16*(j/8) + 8*(lane/16) + (j%8).  With data stored contiguously in k,
// a fragment is two contiguous 8-element (16B) runs at +half8 and +half8+16.
DEV v16bf frag2(const bf16* lo, const bf16* hi) {
  union { v16bf v; v8bf h[2]; } u;
  u.h[0] = *(const v8bf*)lo;
  u.h[1] = *(const v8bf*)hi;
  return u.v;
}

// ---------------------------------------------------------------------------
// TDM: 1-row tensor_load_to_lds of `nelem` bf16 elements (contiguous copy).
// D# per ISA 8.3/8.4: group0 = {count=1, lds_addr, global_addr, type=2},
// group1 = {data_size=2B, tensor_dim0=nelem, tensor_dim1=1, tile_dim0=nelem}.
// ---------------------------------------------------------------------------
DEV void tdm_load_1d(uint32_t lds_byte_addr, const void* gptr, uint32_t nelem) {
  uint64_t ga = (uint64_t)(uintptr_t)gptr;
  v4u g0;
  g0[0] = 1u;                                   // count=1, user mode
  g0[1] = lds_byte_addr;                        // lds_addr
  g0[2] = (uint32_t)ga;                         // global_addr[31:0]
  g0[3] = (uint32_t)((ga >> 32) & 0x1FFFFFFu) | (2u << 30);  // [56:32] | type=2
  v8i g1;
  g1[0] = (1 << 16);                            // data_size=1 (2 bytes)
  g1[1] = (int)((nelem & 0xFFFFu) << 16);       // tensor_dim0[15:0]
  g1[2] = (int)((nelem >> 16) & 0xFFFFu) | (1 << 16);  // dim0[31:16] | dim1=1
  g1[3] = (int)((nelem & 0xFFFFu) << 16);       // tile_dim0 (<= 65535)
  g1[4] = 0;                                    // tile_dim1=0, tile_dim2=0
  g1[5] = (int)nelem;                           // tensor_dim0_stride (unused)
  g1[6] = 0;
  g1[7] = 0;
  v4i z4 = {0, 0, 0, 0};
#if defined(__clang_major__) && (__clang_major__ >= 23)
  v8i z8 = {0, 0, 0, 0, 0, 0, 0, 0};
  __builtin_amdgcn_tensor_load_to_lds(g0, g1, z4, z4, z8, 0);
#else
  __builtin_amdgcn_tensor_load_to_lds(g0, g1, z4, z4, 0);
#endif
}

// ---------------------------------------------------------------------------
// K0: convert fp32 weights -> bf16 in fragment-native swizzled layout:
//   dst[(k/32)*(N*32) + n*32 + (k%32)] = w[k*N + n]
// ---------------------------------------------------------------------------
__global__ void k_cvt_weights(const float* __restrict__ w_in,
                              const float* __restrict__ w_out,
                              bf16* __restrict__ win_swz,
                              bf16* __restrict__ wout_swz) {
  int i = blockIdx.x * 256 + threadIdx.x;
  if (i < DM * N2) {
    int k = i / N2, n = i - k * N2;
    win_swz[(k >> 5) * (N2 * 32) + n * 32 + (k & 31)] = (bf16)w_in[i];
  } else {
    int j = i - DM * N2;
    if (j < DI * DM) {
      int k = j / DM, n = j - k * DM;
      wout_swz[(k >> 5) * (DM * 32) + n * 32 + (k & 31)] = (bf16)w_out[j];
    }
  }
}

// ---------------------------------------------------------------------------
// K1: x_proj = x_flat @ w_in -> x_inner (bf16), silu(z) (bf16)
// 256 threads / 8 waves; tile M=16, N=768; K=192 in 6 steps of 32.
// B slabs (48KB, swizzled) TDM-prefetched into double-buffered LDS.
// ---------------------------------------------------------------------------
__global__ __launch_bounds__(256) void k_gemm_in(
    const float* __restrict__ x, const bf16* __restrict__ win_swz,
    bf16* __restrict__ xin_bf, bf16* __restrict__ z_bf) {
  __shared__ alignas(16) bf16 sA[16 * DM];        // 6 KB
  __shared__ alignas(16) bf16 sB[2 * 32 * N2];    // 2 x 48 KB, [n*32 + k%32]

  const int tid   = threadIdx.x;
  const int lane  = tid & 31;
  const int wave  = tid >> 5;
  const int half8 = (lane >> 4) << 3;
  const int r0    = blockIdx.x * 16;

  // Stage + convert A tile (3072 contiguous floats) as float4
  {
    const float4* src = (const float4*)(x + (size_t)r0 * DM);
    for (int i = tid; i < (16 * DM) / 4; i += 256) {
      float4 v = src[i];
      sA[i * 4 + 0] = (bf16)v.x;  sA[i * 4 + 1] = (bf16)v.y;
      sA[i * 4 + 2] = (bf16)v.z;  sA[i * 4 + 3] = (bf16)v.w;
    }
  }

  // Prefetch slab 0 via TDM
  if (wave == 0)
    tdm_load_1d((uint32_t)(uintptr_t)&sB[0], win_swz, 32 * N2);

  v8f acc[6];
#pragma unroll
  for (int t = 0; t < 6; ++t)
#pragma unroll
    for (int r = 0; r < 8; ++r) acc[t][r] = 0.f;

  for (int s = 0; s < 6; ++s) {
    if (wave == 0) {
      if (s + 1 < 6) {
        tdm_load_1d((uint32_t)(uintptr_t)&sB[((s + 1) & 1) * (32 * N2)],
                    win_swz + (size_t)(s + 1) * (32 * N2), 32 * N2);
        __builtin_amdgcn_s_wait_tensorcnt(1);  // slab s landed
      } else {
        __builtin_amdgcn_s_wait_tensorcnt(0);
      }
    }
    __syncthreads();

    const int k0 = s * 32;
    const bf16* Bb = &sB[(s & 1) * (32 * N2)];
    const bf16* ap = &sA[(lane & 15) * DM + k0 + half8];
    v16bf a = frag2(ap, ap + 16);
#pragma unroll
    for (int t = 0; t < 6; ++t) {
      const bf16* bp = &Bb[(wave * 96 + t * 16 + (lane & 15)) * 32 + half8];
      v16bf b = frag2(bp, bp + 16);
      acc[t] = __builtin_amdgcn_wmma_f32_16x16x32_bf16(
          false, a, false, b, (short)0, acc[t], false, false);
    }
    __syncthreads();  // all reads of slab s done before TDM overwrites it
  }

  // Epilogue: transpose through LDS (reuse sB), coalesced dword stores.
  bf16* sOut = &sB[0];  // 16 x 768 bf16 tile
  const int mbase = (lane >> 4) << 3;
#pragma unroll
  for (int t = 0; t < 6; ++t) {
    const int n = wave * 96 + t * 16 + (lane & 15);
#pragma unroll
    for (int r = 0; r < 8; ++r) {
      float v = acc[t][r];
      sOut[(mbase + r) * N2 + n] = (bf16)((n < DI) ? v : siluf(v));
    }
  }
  __syncthreads();
  {
    const uint32_t* sp = (const uint32_t*)sOut;
    uint32_t* dxin = (uint32_t*)xin_bf;
    uint32_t* dz   = (uint32_t*)z_bf;
    for (int i = tid; i < 16 * (N2 / 2); i += 256) {
      int row = i / (N2 / 2);
      int c2  = i - row * (N2 / 2);        // dword column, half = DI/2 = 192
      uint32_t v = sp[i];
      size_t base = (size_t)(r0 + row) * (DI / 2);
      if (c2 < DI / 2) dxin[base + c2] = v;
      else             dz[base + (c2 - DI / 2)] = v;
    }
  }
}

// ---------------------------------------------------------------------------
// K2: fused conv3x3-dw + SSM + LayerNorm + out-GEMM (WMMA) + transposed store
// Block handles 16 consecutive l's (one h-row segment). 256 threads, 8 waves.
// Halo strips + z tile loaded via TDM (contiguous bf16 runs).
// ---------------------------------------------------------------------------
__global__ __launch_bounds__(256) void k_fused_ssm(
    const bf16* __restrict__ xin_bf, const bf16* __restrict__ z_bf,
    const float* __restrict__ w_conv, const float* __restrict__ b_conv,
    const float* __restrict__ w_x, const float* __restrict__ w_dt,
    const float* __restrict__ b_dt, const float* __restrict__ Dskip,
    const bf16* __restrict__ wout_swz, const float* __restrict__ gamma,
    const float* __restrict__ beta, float* __restrict__ out) {
  __shared__ alignas(16) bf16  s_xi[3 * 18 * DI];  // halo strips
  __shared__ alignas(16) float s_xc[16 * DI];      // xc, reused in-place for y
  __shared__ alignas(16) bf16  s_z [16 * DI];      // silu(z)
  __shared__ alignas(16) float s_xs[16 * 32];      // [B_ssm | C_ssm]
  __shared__             float s_bc[16];
  __shared__ alignas(16) bf16  s_y [16 * DI];      // normalized y (WMMA A)

  const int tid   = threadIdx.x;
  const int lane  = tid & 31;
  const int wave  = tid >> 5;
  const int half8 = (lane >> 4) << 3;
  const int r0    = blockIdx.x * 16;
  const int b     = r0 / LL;
  const int l0    = r0 % LL;
  const int h     = l0 / WWI;
  const int w0    = l0 % WWI;
  const bool wl   = (w0 == 0);
  const bool wr   = (w0 == WWI - 16);

  // Phase 1: TDM halo strips (clip at image edges) + z tile
  if (wave == 0) {
    for (int s = 0; s < 3; ++s) {
      const int hh = h - 1 + s;
      if (hh >= 0 && hh < HH) {
        const int wi0  = wl ? 1 : 0;
        const int wcnt = 18 - wi0 - (wr ? 1 : 0);
        const bf16* src =
            xin_bf + (size_t)(b * LL + hh * WWI + (w0 - 1 + wi0)) * DI;
        tdm_load_1d((uint32_t)(uintptr_t)&s_xi[(s * 18 + wi0) * DI], src,
                    (uint32_t)(wcnt * DI));
      }
    }
    tdm_load_1d((uint32_t)(uintptr_t)&s_z[0], z_bf + (size_t)r0 * DI, 16 * DI);
    __builtin_amdgcn_s_wait_tensorcnt(0);
  }
  __syncthreads();
  // Edge fix-up: zero invalid h-strips and w-edge columns
  for (int s = 0; s < 3; ++s) {
    const int hh = h - 1 + s;
    if (hh < 0 || hh >= HH)
      for (int i = tid; i < 18 * DI; i += 256) s_xi[s * 18 * DI + i] = (bf16)0.f;
  }
  if (wl) for (int i = tid; i < 3 * DI; i += 256)
    s_xi[(i / DI) * (18 * DI) + (i % DI)] = (bf16)0.f;
  if (wr) for (int i = tid; i < 3 * DI; i += 256)
    s_xi[(i / DI) * (18 * DI) + 17 * DI + (i % DI)] = (bf16)0.f;
  __syncthreads();

  // Phase 2: depthwise 3x3 conv + bias + SiLU -> xc (fp32)
  for (int o = tid; o < 16 * DI; o += 256) {
    int m = o / DI, ci = o - m * DI;
    float a = b_conv[ci];
#pragma unroll
    for (int dy = 0; dy < 3; ++dy)
#pragma unroll
      for (int dx = 0; dx < 3; ++dx)
        a += w_conv[ci * 9 + dy * 3 + dx] *
             (float)s_xi[(dy * 18 + m + dx) * DI + ci];
    s_xc[o] = siluf(a);
  }
  __syncthreads();

  // Phase 3: x_ssm = xc @ w_x   (16 x 384) x (384 x 32)
  for (int o = tid; o < 16 * 32; o += 256) {
    int m = o >> 5, j = o & 31;
    float a = 0.f;
    for (int k = 0; k < DI; ++k) a += s_xc[m * DI + k] * w_x[k * 32 + j];
    s_xs[o] = a;
  }
  __syncthreads();

  if (tid < 16) {
    float a = 0.f;
#pragma unroll
    for (int s = 0; s < DS; ++s) a += s_xs[tid * 32 + s] * s_xs[tid * 32 + 16 + s];
    s_bc[tid] = a;
  }
  __syncthreads();

  // Phase 4: y = (softplus(B_ssm@w_dt + b_dt)*bc + xc*D_skip) * silu(z)
  for (int o = tid; o < 16 * DI; o += 256) {
    int m = o / DI, ci = o - m * DI;
    float lin = b_dt[ci];
#pragma unroll
    for (int s = 0; s < DS; ++s) lin += s_xs[m * 32 + s] * w_dt[s * DI + ci];
    float dt = softplusf(lin);
    float y  = dt * s_bc[m] + s_xc[o] * Dskip[ci];
    y *= (float)s_z[o];
    s_xc[o] = y;
  }
  __syncthreads();

  // Phase 5: LayerNorm; wave w handles rows 2w, 2w+1 via wave32 shuffles
#pragma unroll
  for (int rr = 0; rr < 2; ++rr) {
    const int m = wave * 2 + rr;
    float sum = 0.f, sq = 0.f;
#pragma unroll
    for (int i = 0; i < DI / 32; ++i) {
      float y = s_xc[m * DI + lane + 32 * i];
      sum += y; sq += y * y;
    }
#pragma unroll
    for (int d = 16; d > 0; d >>= 1) {
      sum += __shfl_xor(sum, d, 32);
      sq  += __shfl_xor(sq,  d, 32);
    }
    const float mu  = sum * (1.f / DI);
    const float var = sq * (1.f / DI) - mu * mu;
    const float rs  = rsqrtf(var + 1e-5f);
#pragma unroll
    for (int i = 0; i < DI / 32; ++i) {
      const int ci = lane + 32 * i;
      float y = (s_xc[m * DI + ci] - mu) * rs * gamma[ci] + beta[ci];
      s_y[m * DI + ci] = (bf16)y;
    }
  }
  __syncthreads();

  // Phase 6: out = y @ w_out (16x384)x(384x192) via WMMA bf16, swizzled B.
  for (int t = wave; t < 12; t += 8) {
    v8f acc;
#pragma unroll
    for (int r = 0; r < 8; ++r) acc[r] = 0.f;
    for (int k0 = 0; k0 < DI; k0 += 32) {
      const bf16* ap = &s_y[(lane & 15) * DI + k0 + half8];
      const bf16* bp = &wout_swz[(size_t)(k0 >> 5) * (DM * 32) +
                                 (t * 16 + (lane & 15)) * 32 + half8];
      v16bf a  = frag2(ap, ap + 16);
      v16bf bb = frag2(bp, bp + 16);
      acc = __builtin_amdgcn_wmma_f32_16x16x32_bf16(
          false, a, false, bb, (short)0, acc, false, false);
    }
    // Transposed store out[b, ch, l]: 2 x float4 contiguous per lane
    const int ch = t * 16 + (lane & 15);
    const int mb = (lane >> 4) << 3;
    float* dst = out + ((size_t)(b * DM + ch)) * LL + l0 + mb;
    *(float4*)(dst + 0) = make_float4(acc[0], acc[1], acc[2], acc[3]);
    *(float4*)(dst + 4) = make_float4(acc[4], acc[5], acc[6], acc[7]);
  }
}

// ---------------------------------------------------------------------------
extern "C" void kernel_launch(void* const* d_in, const int* in_sizes, int n_in,
                              void* d_out, int out_size, void* d_ws,
                              size_t ws_size, hipStream_t stream) {
  (void)in_sizes; (void)n_in; (void)out_size; (void)ws_size;
  const float* x      = (const float*)d_in[0];
  const float* w_in   = (const float*)d_in[1];
  const float* w_conv = (const float*)d_in[2];
  const float* b_conv = (const float*)d_in[3];
  const float* w_x    = (const float*)d_in[4];
  const float* w_dt   = (const float*)d_in[5];
  const float* b_dt   = (const float*)d_in[6];
  const float* Dskip  = (const float*)d_in[7];
  const float* w_out  = (const float*)d_in[8];
  const float* gamma  = (const float*)d_in[9];
  const float* beta   = (const float*)d_in[10];
  float* out = (float*)d_out;

  char* ws = (char*)d_ws;
  const size_t off_win  = 0;                               // 294912 B
  const size_t off_wout = off_win + (size_t)DM * N2 * 2;   // +147456 B
  const size_t off_xin  = off_wout + (size_t)DI * DM * 2;
  const size_t off_z    = off_xin + (size_t)NROW * DI * 2;
  bf16* win_swz  = (bf16*)(ws + off_win);
  bf16* wout_swz = (bf16*)(ws + off_wout);
  bf16* xin_bf   = (bf16*)(ws + off_xin);
  bf16* z_bf     = (bf16*)(ws + off_z);

  const int cvt_elems = DM * N2 + DI * DM;
  k_cvt_weights<<<(cvt_elems + 255) / 256, 256, 0, stream>>>(
      w_in, w_out, win_swz, wout_swz);
  k_gemm_in<<<NROW / 16, 256, 0, stream>>>(x, win_swz, xin_bf, z_bf);
  k_fused_ssm<<<NROW / 16, 256, 0, stream>>>(
      xin_bf, z_bf, w_conv, b_conv, w_x, w_dt, b_dt, Dskip, wout_swz,
      gamma, beta, out);
}
